// MultiHeadAttention_43533788512646
// MI455X (gfx1250) — matrix operand order repaired and checked by reading, compiled, not run on hardware
//
#include <hip/hip_runtime.h>

// ---------------------------------------------------------------------------
// MHA forward for MI455X (gfx1250): bf16 WMMA pipeline, double-buffered LDS
// staging (async global->LDS via gfx1250 GLOBAL_LOAD_ASYNC_TO_LDS_B128).
//   cvt f32->bf16 -> 3x GEMM (Q/K/V, per-head layout) -> flash attention
//   -> output GEMM (f32 out)
// WMMA fragment layouts follow CDNA5 ISA 7.12.2 (wave32):
//   A 16x32 bf16: lane = m + 16*((k>>3)&1), half = (k&7) + 8*(k>>4)
//   B 32x16 bf16: lane = n + 16*((k>>3)&1), half = (k&7) + 8*(k>>4)
//   C 16x16 f32 : lane = n + 16*(m>>3),     vgpr = m & 7
// ---------------------------------------------------------------------------

typedef __attribute__((ext_vector_type(16))) __bf16 v16bf;
typedef __attribute__((ext_vector_type(4)))  __bf16 v4bf;
typedef __attribute__((ext_vector_type(8)))  float  v8f;
typedef int v4i_ __attribute__((vector_size(16)));   // matches builtin param type

#define BATCH   2
#define SEQ     2048
#define DMODEL  1024
#define NHEADS  16
#define DHEAD   64
#define MROWS   (BATCH * SEQ)          // 4096
#define ATTN_NEG_INF (-1000000000.0f)

// ---- optional gfx1250 async global->LDS path (probe-guarded) --------------
#if defined(__has_builtin)
#  if __has_builtin(__builtin_amdgcn_global_load_async_to_lds_b128) && \
      __has_builtin(__builtin_amdgcn_s_wait_asynccnt)
#    define USE_ASYNC 1
#  endif
#endif
#ifndef USE_ASYNC
#  define USE_ASYNC 0
#endif

__device__ __forceinline__ void cp16_g2l(const __bf16* g, __bf16* l) {
#if USE_ASYNC
  // per-lane 16B copy, tracked by ASYNCcnt
  __builtin_amdgcn_global_load_async_to_lds_b128(
      (__attribute__((address_space(1))) v4i_*)(g),
      (__attribute__((address_space(3))) v4i_*)(l), 0, 0);
#else
  *(float4*)l = *(const float4*)g;
#endif
}
__device__ __forceinline__ void wait_staging() {
#if USE_ASYNC
  __builtin_amdgcn_s_wait_asynccnt(0);
#endif
}

__device__ __forceinline__ int a_frag_off(int m, int k) {
  return ((m & 15) + 16 * ((k >> 3) & 1)) * 16 + ((k & 7) + 8 * ((k >> 4) & 1));
}
__device__ __forceinline__ int b_frag_off(int k, int n) {
  return ((n & 15) + 16 * ((k >> 3) & 1)) * 16 + ((k & 7) + 8 * ((k >> 4) & 1));
}

__device__ __forceinline__ v8f wmma_bf16(v16bf a, v16bf b, v8f c) {
  return __builtin_amdgcn_wmma_f32_16x16x32_bf16(
      false, a, false, b, (short)0, c, false, false);
}

// 4x4 register transpose helper: rows r[0..3] each hold 4 halves (cols c0..c3)
// packed in uint2; returns column c packed as 4 halves (k ascending).
__device__ __forceinline__ uint2 xpose_col(const uint2* r, int c) {
  unsigned h0 = (((c & 2) ? r[0].y : r[0].x) >> ((c & 1) * 16)) & 0xffffu;
  unsigned h1 = (((c & 2) ? r[1].y : r[1].x) >> ((c & 1) * 16)) & 0xffffu;
  unsigned h2 = (((c & 2) ? r[2].y : r[2].x) >> ((c & 1) * 16)) & 0xffffu;
  unsigned h3 = (((c & 2) ? r[3].y : r[3].x) >> ((c & 1) * 16)) & 0xffffu;
  return make_uint2(h0 | (h1 << 16), h2 | (h3 << 16));
}

// ---------------------------------------------------------------------------
// fp32 -> bf16 conversion, float4 vectorized (n multiple of 4)
// ---------------------------------------------------------------------------
__global__ void mha_cvt_f32_bf16(const float* __restrict__ in,
                                 __bf16* __restrict__ out, int n4) {
  int i = blockIdx.x * blockDim.x + threadIdx.x;
  int stride = gridDim.x * blockDim.x;
  for (; i < n4; i += stride) {
    float4 v = *(const float4*)(in + (size_t)i * 4);
    v4bf o = {(__bf16)v.x, (__bf16)v.y, (__bf16)v.z, (__bf16)v.w};
    *(v4bf*)(out + (size_t)i * 4) = o;
  }
}

// ---------------------------------------------------------------------------
// 128x128 tile GEMM:  C[M,N] = A[M,K] @ B[K,N] + bias, double-buffered LDS.
//   MODE 0: write bf16 to per-head layout [B, NHEADS, L, DHEAD]
//   MODE 1: write f32 row-major [M, N]
// ---------------------------------------------------------------------------
template <int MODE>
__global__ __launch_bounds__(256)
void mha_gemm_bf16_128x128(const __bf16* __restrict__ A,
                           const __bf16* __restrict__ Bm,
                           const float*  __restrict__ bias,
                           void* __restrict__ Cout,
                           int M, int N, int K, int L) {
  __shared__ __align__(32) __bf16 lA[2][8 * 512];
  __shared__ __align__(32) __bf16 lB[2][8 * 512];

  const int tid  = threadIdx.x;
  const int lane = tid & 31;
  const int wave = tid >> 5;
  const int wm   = wave & 1;
  const int wn   = wave >> 1;
  const int mBase = blockIdx.y * 128;
  const int nBase = blockIdx.x * 128;

  // staging thread maps (hoisted)
  const int a_m0 = (tid * 8) >> 5,            a_k0 = (tid * 8) & 31;
  const int a_m1 = ((tid + 256) * 8) >> 5,    a_k1 = ((tid + 256) * 8) & 31;
  const int b_n0 = (tid & 31) * 4,            b_k0 = (tid >> 5) * 4;

  auto stage = [&](int buf, int kb) {
    // A tile 128x32 -> fragment layout, 16B contiguous both sides
    cp16_g2l(A + (size_t)(mBase + a_m0) * K + kb + a_k0,
             &lA[buf][(a_m0 >> 4) * 512 + a_frag_off(a_m0 & 15, a_k0)]);
    cp16_g2l(A + (size_t)(mBase + a_m1) * K + kb + a_k1,
             &lA[buf][(a_m1 >> 4) * 512 + a_frag_off(a_m1 & 15, a_k1)]);
    // B tile 32x128 -> fragment layout via 4x4 register transpose
    uint2 r[4];
#pragma unroll
    for (int rr = 0; rr < 4; ++rr)
      r[rr] = *(const uint2*)(Bm + (size_t)(kb + b_k0 + rr) * N + nBase + b_n0);
#pragma unroll
    for (int c = 0; c < 4; ++c) {
      int n = b_n0 + c;
      *(uint2*)(&lB[buf][(n >> 4) * 512 + b_frag_off(b_k0, n & 15)]) =
          xpose_col(r, c);
    }
  };

  v8f acc[4][2];
#pragma unroll
  for (int i = 0; i < 4; ++i)
#pragma unroll
    for (int j = 0; j < 2; ++j) acc[i][j] = {};

  const int nk = K / 32;
  stage(0, 0);
  wait_staging();
  __syncthreads();

  for (int kt = 0; kt < nk; ++kt) {
    const int cur = kt & 1;
    if (kt + 1 < nk) stage(cur ^ 1, (kt + 1) * 32);

    v16bf af[4], bfr[2];
#pragma unroll
    for (int i = 0; i < 4; ++i)
      af[i] = *(const v16bf*)(&lA[cur][(wm * 4 + i) * 512 + lane * 16]);
#pragma unroll
    for (int j = 0; j < 2; ++j)
      bfr[j] = *(const v16bf*)(&lB[cur][(wn * 2 + j) * 512 + lane * 16]);
#pragma unroll
    for (int i = 0; i < 4; ++i)
#pragma unroll
      for (int j = 0; j < 2; ++j) acc[i][j] = wmma_bf16(af[i], bfr[j], acc[i][j]);

    wait_staging();
    __syncthreads();
  }

  // ---- branch-free epilogue ----
  const int col = lane & 15, rg = lane >> 4;
#pragma unroll
  for (int i = 0; i < 4; ++i) {
#pragma unroll
    for (int j = 0; j < 2; ++j) {
      int n = nBase + wn * 32 + j * 16 + col;
      float bv = bias[n];
#pragma unroll
      for (int v = 0; v < 8; ++v) {
        int m = mBase + wm * 64 + i * 16 + v + 8 * rg;
        float val = acc[i][j][v] + bv;
        if (MODE == 0) {
          int bidx = m / L, l = m % L, hh = n >> 6, d = n & 63;
          ((__bf16*)Cout)[(((size_t)bidx * NHEADS + hh) * L + l) * DHEAD + d] =
              (__bf16)val;
        } else {
          ((float*)Cout)[(size_t)m * N + n] = val;
        }
      }
    }
  }
}

// ---------------------------------------------------------------------------
// Flash attention, one (b,h) x 64-q-row block per workgroup (4 waves),
// double-buffered K/V chunks of 64 keys.
// ---------------------------------------------------------------------------
__global__ __launch_bounds__(128)
void mha_flash_attn_bf16(const __bf16* __restrict__ Q,
                         const __bf16* __restrict__ Kt,
                         const __bf16* __restrict__ Vt,
                         const float*  __restrict__ mask,
                         __bf16* __restrict__ ctx) {
  __shared__ __align__(32) __bf16 lQ[8 * 512];
  __shared__ __align__(32) __bf16 lK[2][8 * 512];
  __shared__ __align__(32) __bf16 lV[2][8 * 512];
  __shared__ __align__(32) __bf16 lP[4][2 * 512];

  const int tid = threadIdx.x, lane = tid & 31, wave = tid >> 5;
  const int col = lane & 15, rg = lane >> 4;
  const int bh = blockIdx.y;
  const int b  = bh >> 4;
  const int h  = bh & 15;
  const int qbase = blockIdx.x * 64;
  const size_t seqbase = (size_t)bh * SEQ;

  auto stageKV = [&](int buf, int kc) {
    // K chunk (64 keys x 64 d) -> B-frags for S=QK^T: 16B contiguous
#pragma unroll
    for (int i = 0; i < 4; ++i) {
      int linear = (tid + i * 128) * 8;
      int key = linear >> 6, d = linear & 63;
      cp16_g2l(Kt + (seqbase + kc + key) * DHEAD + d,
               &lK[buf][((d >> 5) * 4 + (key >> 4)) * 512 +
                        b_frag_off(d & 31, key & 15)]);
    }
    // V chunk -> B-frags for O=PV via 4x4 register transpose (2 blocks/thread)
#pragma unroll
    for (int i = 0; i < 2; ++i) {
      int blk = tid + i * 128;
      int d0 = (blk & 15) * 4, key0 = (blk >> 4) * 4;
      uint2 r[4];
#pragma unroll
      for (int rr = 0; rr < 4; ++rr)
        r[rr] = *(const uint2*)(Vt + (seqbase + kc + key0 + rr) * DHEAD + d0);
#pragma unroll
      for (int c = 0; c < 4; ++c) {
        int d = d0 + c;
        *(uint2*)(&lV[buf][((key0 >> 5) * 4 + (d >> 4)) * 512 +
                           b_frag_off(key0 & 31, d & 15)]) = xpose_col(r, c);
      }
    }
  };

  // ---- stage Q block (64x64) into A-fragment layout, once ----
#pragma unroll
  for (int i = 0; i < 4; ++i) {
    int linear = (tid + i * 128) * 8;
    int r = linear >> 6, d = linear & 63;
    cp16_g2l(Q + (seqbase + qbase + r) * DHEAD + d,
             &lQ[((r >> 4) * 2 + (d >> 5)) * 512 + a_frag_off(r & 15, d & 31)]);
  }
  stageKV(0, 0);
  wait_staging();
  __syncthreads();

  v16bf qf[2];
  qf[0] = *(const v16bf*)(&lQ[(wave * 2 + 0) * 512 + lane * 16]);
  qf[1] = *(const v16bf*)(&lQ[(wave * 2 + 1) * 512 + lane * 16]);

  v8f acc[4];
#pragma unroll
  for (int i = 0; i < 4; ++i) acc[i] = {};
  float rowm[8], rowl[8];
#pragma unroll
  for (int v = 0; v < 8; ++v) { rowm[v] = -3.0e38f; rowl[v] = 0.0f; }

  const int nchunks = SEQ / 64;
  for (int ic = 0; ic < nchunks; ++ic) {
    const int cur = ic & 1;
    const int kc = ic * 64;
    if (ic + 1 < nchunks) stageKV(cur ^ 1, kc + 64);

    // ---- scores ----
    v8f s[4];
#pragma unroll
    for (int nf = 0; nf < 4; ++nf) {
      v8f t = {};
      t = wmma_bf16(qf[0], *(const v16bf*)(&lK[cur][(0 * 4 + nf) * 512 + lane * 16]), t);
      t = wmma_bf16(qf[1], *(const v16bf*)(&lK[cur][(1 * 4 + nf) * 512 + lane * 16]), t);
      s[nf] = t;
    }
    // ---- scale, multiplicative mask, where(==0, -1e9) ----
#pragma unroll
    for (int nf = 0; nf < 4; ++nf) {
      int kcol = kc + nf * 16 + col;
#pragma unroll
      for (int v = 0; v < 8; ++v) {
        int qrow = qbase + wave * 16 + v + 8 * rg;
        float mv = mask[((size_t)b * SEQ + qrow) * SEQ + kcol];
        float sv = s[nf][v] * 0.125f * mv;
        if (sv == 0.0f) sv = ATTN_NEG_INF;
        s[nf][v] = sv;
      }
    }
    // ---- online softmax (row lives in one VGPR across a 16-lane group) ----
    float p[4][8];
#pragma unroll
    for (int v = 0; v < 8; ++v) {
      float mx = fmaxf(fmaxf(s[0][v], s[1][v]), fmaxf(s[2][v], s[3][v]));
#pragma unroll
      for (int o = 1; o < 16; o <<= 1) mx = fmaxf(mx, __shfl_xor(mx, o, 32));
      float mnew  = fmaxf(rowm[v], mx);
      float scale = __expf(rowm[v] - mnew);
      rowm[v] = mnew;
      float rs = 0.0f;
#pragma unroll
      for (int nf = 0; nf < 4; ++nf) {
        p[nf][v] = __expf(s[nf][v] - mnew);
        rs += p[nf][v];
      }
#pragma unroll
      for (int o = 1; o < 16; o <<= 1) rs += __shfl_xor(rs, o, 32);
      rowl[v] = rowl[v] * scale + rs;
#pragma unroll
      for (int nf = 0; nf < 4; ++nf) acc[nf][v] *= scale;
    }
    // ---- repack P (C layout) -> A-fragment layout via per-wave LDS ----
#pragma unroll
    for (int nf = 0; nf < 4; ++nf) {
#pragma unroll
      for (int v = 0; v < 8; ++v) {
        int r = v + 8 * rg;
        int k = nf * 16 + col;
        lP[wave][(k >> 5) * 512 + a_frag_off(r, k & 31)] = (__bf16)p[nf][v];
      }
    }
    v16bf pf0 = *(const v16bf*)(&lP[wave][0 * 512 + lane * 16]);
    v16bf pf1 = *(const v16bf*)(&lP[wave][1 * 512 + lane * 16]);
#pragma unroll
    for (int nf = 0; nf < 4; ++nf) {
      acc[nf] = wmma_bf16(pf0, *(const v16bf*)(&lV[cur][(0 * 4 + nf) * 512 + lane * 16]), acc[nf]);
      acc[nf] = wmma_bf16(pf1, *(const v16bf*)(&lV[cur][(1 * 4 + nf) * 512 + lane * 16]), acc[nf]);
    }

    wait_staging();
    __syncthreads();
  }

  // ---- normalize and write context [B, SEQ, DMODEL] ----
#pragma unroll
  for (int nf = 0; nf < 4; ++nf) {
#pragma unroll
    for (int v = 0; v < 8; ++v) {
      int qrow = qbase + wave * 16 + v + 8 * rg;
      int dcol = h * DHEAD + nf * 16 + col;
      float val = acc[nf][v] / rowl[v];
      ctx[((size_t)b * SEQ + qrow) * DMODEL + dcol] = (__bf16)val;
    }
  }
}

// ---------------------------------------------------------------------------
// Host launcher
// ---------------------------------------------------------------------------
extern "C" void kernel_launch(void* const* d_in, const int* in_sizes, int n_in,
                              void* d_out, int out_size, void* d_ws, size_t ws_size,
                              hipStream_t stream) {
  const float* dec   = (const float*)d_in[0];
  const float* enc   = (const float*)d_in[1];
  const float* mask  = (const float*)d_in[2];
  const float* wq_w  = (const float*)d_in[3];
  const float* wq_b  = (const float*)d_in[4];
  const float* wk_w  = (const float*)d_in[5];
  const float* wk_b  = (const float*)d_in[6];
  const float* wv_w  = (const float*)d_in[7];
  const float* wv_b  = (const float*)d_in[8];
  const float* wo_w  = (const float*)d_in[9];
  const float* wo_b  = (const float*)d_in[10];
  float* out = (float*)d_out;

  char* ws = (char*)d_ws;
  size_t off = 0;
  auto alloc = [&](size_t bytes) -> char* {
    char* p = ws + off;
    off += (bytes + 255) & ~(size_t)255;
    return p;
  };
  const size_t actN = (size_t)MROWS * DMODEL;
  const size_t wN   = (size_t)DMODEL * DMODEL;

  __bf16* decb = (__bf16*)alloc(actN * 2);
  __bf16* encb = (__bf16*)alloc(actN * 2);
  __bf16* wqb  = (__bf16*)alloc(wN * 2);
  __bf16* wkb  = (__bf16*)alloc(wN * 2);
  __bf16* wvb  = (__bf16*)alloc(wN * 2);
  __bf16* wob  = (__bf16*)alloc(wN * 2);
  __bf16* Qb   = (__bf16*)alloc(actN * 2);
  __bf16* Kb   = (__bf16*)alloc(actN * 2);
  __bf16* Vb   = (__bf16*)alloc(actN * 2);
  __bf16* ctxb = (__bf16*)alloc(actN * 2);

  auto cvt = [&](const float* i, __bf16* o, size_t n) {
    int n4 = (int)(n / 4);
    mha_cvt_f32_bf16<<<(n4 + 255) / 256, 256, 0, stream>>>(i, o, n4);
  };
  cvt(dec,  decb, actN);
  cvt(enc,  encb, actN);
  cvt(wq_w, wqb,  wN);
  cvt(wk_w, wkb,  wN);
  cvt(wv_w, wvb,  wN);
  cvt(wo_w, wob,  wN);

  dim3 gg(DMODEL / 128, MROWS / 128);  // (8, 32)
  mha_gemm_bf16_128x128<0><<<gg, 256, 0, stream>>>(decb, wqb, wq_b, Qb,
                                                   MROWS, DMODEL, DMODEL, SEQ);
  mha_gemm_bf16_128x128<0><<<gg, 256, 0, stream>>>(encb, wkb, wk_b, Kb,
                                                   MROWS, DMODEL, DMODEL, SEQ);
  mha_gemm_bf16_128x128<0><<<gg, 256, 0, stream>>>(encb, wvb, wv_b, Vb,
                                                   MROWS, DMODEL, DMODEL, SEQ);

  dim3 ga(SEQ / 64, BATCH * NHEADS);   // (32, 32)
  mha_flash_attn_bf16<<<ga, 128, 0, stream>>>(Qb, Kb, Vb, mask, ctxb);

  mha_gemm_bf16_128x128<1><<<gg, 256, 0, stream>>>(ctxb, wob, wo_b, out,
                                                   MROWS, DMODEL, DMODEL, SEQ);
}
